// ZonotopeTransformer_16836271801085
// MI455X (gfx1250) — compile-verified, matrix operand order checked by reading, not run on hardware
//
#include <hip/hip_runtime.h>
#include <hip/hip_bf16.h>

// ---------------------------------------------------------------------------
// Zonotope 3-layer MLP forward for MI455X (gfx1250, wave32, WMMA).
// Heavy GEMMs in bf16 WMMA (v_wmma_f32_16x16x32_bf16, f32 accumulate).
// Wave tile 64x64 (4x4 fragments -> 16 WMMAs per K=32 step, 16 b128 loads),
// block = 4 waves (2x2) -> 128x128 tile. Operands are L2-resident (192 MB L2;
// bf16 A/W total < 80 MB), so direct global loads feed the matrix pipe.
// Diagonal error-row blocks are handled analytically (row scalings / matvecs)
// instead of materializing [4096,4096] blocks.
// ---------------------------------------------------------------------------

typedef __attribute__((ext_vector_type(16))) __bf16 v16bf;
typedef __attribute__((ext_vector_type(8)))  __bf16 v8bf;
typedef __attribute__((ext_vector_type(8)))  float  v8f;

static constexpr int D = 3072;   // input dim / #initial error terms
static constexpr int H = 4096;   // hidden dim
static constexpr int O = 10;     // output dim
static constexpr float MU = 0.5f;
static constexpr float SIGMA = 0.5f;

// ------------------------------ elementwise --------------------------------

__global__ void cast_bf16_kernel(const float* __restrict__ in,
                                 __hip_bfloat16* __restrict__ out,
                                 size_t n, float scale) {
  for (size_t i = blockIdx.x * (size_t)blockDim.x + threadIdx.x; i < n;
       i += (size_t)gridDim.x * blockDim.x)
    out[i] = __float2bfloat16(in[i] * scale);
}

__global__ void affine_kernel(const float* __restrict__ in,
                              float* __restrict__ out, int n,
                              float sub, float mul) {
  int i = blockIdx.x * blockDim.x + threadIdx.x;
  if (i < n) out[i] = (in[i] - sub) * mul;
}

// scale columns by colscale and cast to bf16 (prep for next WMMA GEMM)
__global__ void scale_cast_kernel(const float* __restrict__ Min,
                                  const float* __restrict__ colscale,
                                  __hip_bfloat16* __restrict__ out,
                                  size_t total, int C) {
  for (size_t idx = blockIdx.x * (size_t)blockDim.x + threadIdx.x; idx < total;
       idx += (size_t)gridDim.x * blockDim.x) {
    int n = (int)(idx % (size_t)C);
    out[idx] = __float2bfloat16(Min[idx] * colscale[n]);
  }
}

// ------------------------------ WMMA GEMM ----------------------------------
// C[M,N] = A[M,K] @ B[N,K]^T     (A, B row-major along K, bf16; C f32)
// Wave tile 64x64 (4x4 fragments), block = 4 waves (2x2) -> 128x128 tile.

union Frag { v16bf v; v8bf h[2]; };

__device__ __forceinline__ void load_frag(Frag& f, const __bf16* __restrict__ base,
                                          int row, int ldk, int kb, int hi) {
  const __bf16* p = base + (size_t)row * ldk + kb + hi * 8;
  f.h[0] = *reinterpret_cast<const v8bf*>(p);        // K = kb+8*hi   .. +7
  f.h[1] = *reinterpret_cast<const v8bf*>(p + 16);   // K = kb+16+8*hi.. +7
}

__global__ __launch_bounds__(128) void wmma_gemm_bf16_kernel(
    const __bf16* __restrict__ A, const __bf16* __restrict__ B,
    float* __restrict__ C, int M, int N, int K) {
  const int wave = threadIdx.x >> 5;
  const int lane = threadIdx.x & 31;
  const int r    = lane & 15;
  const int hi   = lane >> 4;
  const int m0 = blockIdx.y * 128 + (wave >> 1) * 64;
  const int n0 = blockIdx.x * 128 + (wave & 1) * 64;

  const v8f vzero = {0.f, 0.f, 0.f, 0.f, 0.f, 0.f, 0.f, 0.f};
  v8f acc[4][4];
#pragma unroll
  for (int i = 0; i < 4; ++i)
#pragma unroll
    for (int j = 0; j < 4; ++j) acc[i][j] = vzero;

  for (int kb = 0; kb < K; kb += 32) {
    Frag a[4], b[4];
#pragma unroll
    for (int i = 0; i < 4; ++i) load_frag(a[i], A, m0 + i * 16 + r, K, kb, hi);
#pragma unroll
    for (int j = 0; j < 4; ++j) load_frag(b[j], B, n0 + j * 16 + r, K, kb, hi);
#pragma unroll
    for (int i = 0; i < 4; ++i)
#pragma unroll
      for (int j = 0; j < 4; ++j)
        acc[i][j] = __builtin_amdgcn_wmma_f32_16x16x32_bf16(
            false, a[i].v, false, b[j].v, (short)0, acc[i][j], false, false);
  }

#pragma unroll
  for (int i = 0; i < 4; ++i)
#pragma unroll
    for (int j = 0; j < 4; ++j) {
      const int rbase = m0 + i * 16 + hi * 8;   // M = v + 8*hi within tile
      const int ncol  = n0 + j * 16 + r;
#pragma unroll
      for (int v = 0; v < 8; ++v)
        C[(size_t)(rbase + v) * N + ncol] = acc[i][j][v];
    }
}

// ------------------------------ reductions ---------------------------------

// out[n] = dot(v, W[n,:]) + b[n]      (one block per n)
__global__ __launch_bounds__(256) void dot_bias_kernel(
    const float* __restrict__ v, const float* __restrict__ W,
    const float* __restrict__ b, float* __restrict__ out, int K) {
  int n = blockIdx.x;
  const float* w = W + (size_t)n * K;
  float s = 0.f;
  for (int k = threadIdx.x; k < K; k += 256) s += v[k] * w[k];
  __shared__ float red[256];
  red[threadIdx.x] = s;
  __syncthreads();
  for (int offc = 128; offc > 0; offc >>= 1) {
    if (threadIdx.x < offc) red[threadIdx.x] += red[threadIdx.x + offc];
    __syncthreads();
  }
  if (threadIdx.x == 0) out[n] = red[0] + b[n];
}

// out[n] = dot(v, |W[n,:]|)           (one block per n)
__global__ __launch_bounds__(256) void absdot_kernel(
    const float* __restrict__ v, const float* __restrict__ W,
    float* __restrict__ out, int K) {
  int n = blockIdx.x;
  const float* w = W + (size_t)n * K;
  float s = 0.f;
  for (int k = threadIdx.x; k < K; k += 256) s += v[k] * fabsf(w[k]);
  __shared__ float red[256];
  red[threadIdx.x] = s;
  __syncthreads();
  for (int offc = 128; offc > 0; offc >>= 1) {
    if (threadIdx.x < offc) red[threadIdx.x] += red[threadIdx.x + offc];
    __syncthreads();
  }
  if (threadIdx.x == 0) out[n] = red[0];
}

// out[n] = sum_i |Mat[i,n]|   (thread per column; coalesced row sweeps)
__global__ void colabs_kernel(const float* __restrict__ Mat,
                              float* __restrict__ out, int R, int C) {
  int n = blockIdx.x * blockDim.x + threadIdx.x;
  if (n >= C) return;
  float s = 0.f;
  for (int i = 0; i < R; ++i) s += fabsf(Mat[(size_t)i * C + n]);
  out[n] = s;
}

// ------------------------------ ReLU-zono stats ----------------------------

__global__ void relu_stats_kernel(const float* __restrict__ c_in,
                                  const float* __restrict__ absA,
                                  const float* __restrict__ absB,
                                  const float* __restrict__ slopes,
                                  float* __restrict__ colscale,
                                  float* __restrict__ mag,
                                  float* __restrict__ c_out, int n) {
  int i = blockIdx.x * blockDim.x + threadIdx.x;
  if (i >= n) return;
  float a = absA[i] + (absB ? absB[i] : 0.f);
  float cc = c_in[i];
  float lx = cc - a, ux = cc + a;
  float s = slopes[i];
  bool zero  = ux <= 0.f;
  bool cross = (ux > 0.f) && (lx <= 0.f);
  float basic = ux / (ux - lx);
  bool in_range = (s >= 0.f) && (s <= basic);
  float mg = cross ? (in_range ? (1.f - s) * ux * 0.5f : (-s * lx * 0.5f)) : 0.f;
  colscale[i] = zero ? 0.f : (cross ? s : 1.f);
  mag[i] = mg;
  c_out[i] = zero ? 0.f : (cross ? (s * cc + mg) : cc);
}

// ------------------------------ layer-3 tail -------------------------------

// rows 0..D-1 of output: out[i,k] = sum_n M2top[i,n]*cs2[n]*W3[k,n]
__global__ __launch_bounds__(256) void out_top_kernel(
    const float* __restrict__ M2, const float* __restrict__ cs2,
    const float* __restrict__ W3, float* __restrict__ out) {
  int i = blockIdx.x;
  float acc[O];
#pragma unroll
  for (int k = 0; k < O; ++k) acc[k] = 0.f;
  for (int n = threadIdx.x; n < H; n += 256) {
    float t = M2[(size_t)i * H + n] * cs2[n];
#pragma unroll
    for (int k = 0; k < O; ++k) acc[k] += t * W3[k * H + n];
  }
  __shared__ float red[256];
  for (int k = 0; k < O; ++k) {
    red[threadIdx.x] = acc[k];
    __syncthreads();
    for (int offc = 128; offc > 0; offc >>= 1) {
      if (threadIdx.x < offc) red[threadIdx.x] += red[threadIdx.x + offc];
      __syncthreads();
    }
    if (threadIdx.x == 0) out[(size_t)i * O + k] = red[0];
    __syncthreads();
  }
}

// rows D..D+H-1: out[D+i,k] = mag1[i] * sum_j W3[k,j]*cs2[j]*W2[j,i]
__global__ __launch_bounds__(256) void mid_kernel(
    const float* __restrict__ W2, const float* __restrict__ W3,
    const float* __restrict__ cs2, const float* __restrict__ mag1,
    float* __restrict__ out) {
  __shared__ float w3s[O * 128];
  int i = blockIdx.x * 256 + threadIdx.x;  // column of W2
  float acc[O];
#pragma unroll
  for (int k = 0; k < O; ++k) acc[k] = 0.f;
  for (int jb = 0; jb < H; jb += 128) {
    for (int idx = threadIdx.x; idx < O * 128; idx += 256) {
      int k = idx >> 7, jj = idx & 127;
      w3s[idx] = W3[k * H + jb + jj] * cs2[jb + jj];
    }
    __syncthreads();
    for (int jj = 0; jj < 128; ++jj) {
      float w2v = W2[(size_t)(jb + jj) * H + i];  // coalesced across threads
#pragma unroll
      for (int k = 0; k < O; ++k) acc[k] += w3s[k * 128 + jj] * w2v;
    }
    __syncthreads();
  }
  float m = mag1[i];
#pragma unroll
  for (int k = 0; k < O; ++k) out[(size_t)(D + i) * O + k] = m * acc[k];
}

// rows D+H..D+2H-1: out[D+H+i,k] = mag2[i]*W3[k,i]
__global__ void tail_kernel(const float* __restrict__ mag2,
                            const float* __restrict__ W3,
                            float* __restrict__ out) {
  int i = blockIdx.x * blockDim.x + threadIdx.x;
  if (i >= H) return;
  float m = mag2[i];
#pragma unroll
  for (int k = 0; k < O; ++k) out[(size_t)(D + H + i) * O + k] = m * W3[k * H + i];
}

// ------------------------------ host orchestration -------------------------

extern "C" void kernel_launch(void* const* d_in, const int* in_sizes, int n_in,
                              void* d_out, int out_size, void* d_ws, size_t ws_size,
                              hipStream_t stream) {
  const float* Mmat   = (const float*)d_in[0];
  const float* cvec   = (const float*)d_in[1];
  const float* slopes1= (const float*)d_in[2];
  const float* slopes2= (const float*)d_in[3];
  const float* W1     = (const float*)d_in[4];
  const float* b1     = (const float*)d_in[5];
  const float* W2     = (const float*)d_in[6];
  const float* b2     = (const float*)d_in[7];
  const float* W3     = (const float*)d_in[8];
  const float* b3     = (const float*)d_in[9];
  float* out = (float*)d_out;

  char* ws = (char*)d_ws;
  size_t off = 0;
  auto alloc = [&](size_t bytes) -> void* {
    void* p = ws + off;
    off = (off + bytes + 255) & ~(size_t)255;
    return p;
  };

  __hip_bfloat16* Mbf   = (__hip_bfloat16*)alloc((size_t)D * D * 2);
  __hip_bfloat16* W1bf  = (__hip_bfloat16*)alloc((size_t)H * D * 2);
  __hip_bfloat16* W2bf  = (__hip_bfloat16*)alloc((size_t)H * H * 2);
  float*          M1    = (float*)alloc((size_t)D * H * 4);  // reused as M2top
  __hip_bfloat16* M1sbf = (__hip_bfloat16*)alloc((size_t)D * H * 2);
  float* cnorm  = (float*)alloc(D * 4);
  float* c1     = (float*)alloc(H * 4);
  float* abs1   = (float*)alloc(H * 4);
  float* cs1    = (float*)alloc(H * 4);
  float* mag1   = (float*)alloc(H * 4);
  float* c1n    = (float*)alloc(H * 4);
  float* c2pre  = (float*)alloc(H * 4);
  float* botabs = (float*)alloc(H * 4);
  float* abs2t  = (float*)alloc(H * 4);
  float* cs2    = (float*)alloc(H * 4);
  float* mag2   = (float*)alloc(H * 4);
  float* c2n    = (float*)alloc(H * 4);
  float* M2top  = M1;  // alias: M1 dead once M1sbf is built

  // ---- prep: bf16 casts + normalized center
  cast_bf16_kernel<<<4096, 256, 0, stream>>>(Mmat, Mbf, (size_t)D * D, 1.f / SIGMA);
  cast_bf16_kernel<<<4096, 256, 0, stream>>>(W1, W1bf, (size_t)H * D, 1.f);
  cast_bf16_kernel<<<4096, 256, 0, stream>>>(W2, W2bf, (size_t)H * H, 1.f);
  affine_kernel<<<(D + 255) / 256, 256, 0, stream>>>(cvec, cnorm, D, MU, 1.f / SIGMA);

  // ---- layer 1: M1 = (M/sigma) @ W1^T ; c1 = cnorm @ W1^T + b1
  wmma_gemm_bf16_kernel<<<dim3(H / 128, D / 128), 128, 0, stream>>>(
      (const __bf16*)Mbf, (const __bf16*)W1bf, M1, D, H, D);
  dot_bias_kernel<<<H, 256, 0, stream>>>(cnorm, W1, b1, c1, D);

  // ---- relu-zono 1
  colabs_kernel<<<H / 256, 256, 0, stream>>>(M1, abs1, D, H);
  relu_stats_kernel<<<H / 256, 256, 0, stream>>>(c1, abs1, nullptr, slopes1,
                                                 cs1, mag1, c1n, H);
  scale_cast_kernel<<<4096, 256, 0, stream>>>(M1, cs1, M1sbf, (size_t)D * H, H);

  // ---- layer 2: M2top = M1s @ W2^T ; c2 = c1n @ W2^T + b2 ; diag block abs
  wmma_gemm_bf16_kernel<<<dim3(H / 128, D / 128), 128, 0, stream>>>(
      (const __bf16*)M1sbf, (const __bf16*)W2bf, M2top, D, H, H);
  dot_bias_kernel<<<H, 256, 0, stream>>>(c1n, W2, b2, c2pre, H);
  absdot_kernel<<<H, 256, 0, stream>>>(mag1, W2, botabs, H);

  // ---- relu-zono 2 (abs = column-abs of dense block + |W2|^T mag1)
  colabs_kernel<<<H / 256, 256, 0, stream>>>(M2top, abs2t, D, H);
  relu_stats_kernel<<<H / 256, 256, 0, stream>>>(c2pre, abs2t, botabs, slopes2,
                                                 cs2, mag2, c2n, H);

  // ---- layer 3 (N=10): dense rows, diag-block-1 rows, diag-block-2 rows, center
  out_top_kernel<<<D, 256, 0, stream>>>(M2top, cs2, W3, out);
  mid_kernel<<<H / 256, 256, 0, stream>>>(W2, W3, cs2, mag1, out);
  tail_kernel<<<H / 256, 256, 0, stream>>>(mag2, W3, out);
  dot_bias_kernel<<<O, 256, 0, stream>>>(c2n, W3, b3, out + (size_t)(D + 2 * H) * O, H);
}
